// SparseQuery_47313359732999
// MI455X (gfx1250) — matrix operand order repaired and checked by reading, not compiled, not run on hardware
//
#include <hip/hip_runtime.h>
#include <math.h>

// ---- problem constants (from reference) ----
#define DIN   2048
#define HIDN  256
#define NHEAD 16
#define HD    128
#define KSEL  4
#define MTOT  (4 * 4096)          // B*S = 16384 tokens
#define EPSN  1e-12f

typedef __attribute__((ext_vector_type(16))) __bf16 v16bf;
typedef __attribute__((ext_vector_type(8)))  float  v8f;
typedef __attribute__((ext_vector_type(2)))  __bf16 bf16x2;
typedef __attribute__((ext_vector_type(4)))  int    v4i;

union FragAB { v16bf v; float4 f[2]; };

// ---- fp32 -> bf16 packing (targets v_cvt_pk_bf16_f32) ----
__device__ __forceinline__ unsigned pack2(float x, float y) {
    bf16x2 p = {(__bf16)x, (__bf16)y};
    return __builtin_bit_cast(unsigned, p);
}
__device__ __forceinline__ uint2 pack_bf16x4(float4 a) {
    return uint2{pack2(a.x, a.y), pack2(a.z, a.w)};
}

// ---- CDNA5 async global->LDS copy (ASYNCcnt path) ----
// Builtin signature (from hipcc diagnostic): param0 = v4i addrspace(1)* (global),
// param1 = LDS side, then imm offset, imm cpol.
typedef __attribute__((address_space(1))) v4i* as1_v4i_ptr;
typedef __attribute__((address_space(3))) v4i* as3_v4i_ptr;

__device__ __forceinline__ void async_g2l_b128(const void* g, void* l) {
#if __has_builtin(__builtin_amdgcn_global_load_async_to_lds_b128)
    __builtin_amdgcn_global_load_async_to_lds_b128(
        (as1_v4i_ptr)g, (as3_v4i_ptr)l, 0, 0);
#else
    unsigned loff = (unsigned)(__UINTPTR_TYPE__)(__attribute__((address_space(3))) char*)l;
    asm volatile("global_load_async_to_lds_b128 %0, %1, off"
                 :: "v"(loff), "v"((unsigned long long)g) : "memory");
#endif
}

template <int N>
__device__ __forceinline__ void wait_async() {
#if __has_builtin(__builtin_amdgcn_s_wait_asynccnt)
    __builtin_amdgcn_s_wait_asynccnt((short)N);
#else
    asm volatile("s_wait_asynccnt %0" :: "i"(N) : "memory");
#endif
}

// ============================================================================
// Kernel 0: bulk fp32 -> bf16 conversion (8 elems / thread, b128 in, b128 out)
// ============================================================================
__global__ __launch_bounds__(256) void conv_f32_bf16(
    const float* __restrict__ src, __bf16* __restrict__ dst)
{
    const size_t i = (size_t)blockIdx.x * 256 + threadIdx.x;   // one 8-elem chunk
    const float4 a = *reinterpret_cast<const float4*>(src + i * 8);
    const float4 b = *reinterpret_cast<const float4*>(src + i * 8 + 4);
    uint4 o;
    o.x = pack2(a.x, a.y); o.y = pack2(a.z, a.w);
    o.z = pack2(b.x, b.y); o.w = pack2(b.z, b.w);
    *reinterpret_cast<uint4*>(dst + i * 8) = o;
}

// ============================================================================
// Kernel 1: routing.  z = x @ router_w^T (WMMA bf16), L2-normalize, cosine
// logits vs normalized centroids, softmax over 16 heads, top-4 -> ws.
// Block: 256 threads (8 waves), tile M=32 x N=256 (all of HID), K-loop 32.
// ============================================================================
__global__ __launch_bounds__(256) void sq_router(
    const float* __restrict__ x, const float* __restrict__ rw,
    const float* __restrict__ cent, const float* __restrict__ temp,
    int* __restrict__ topk_idx, float* __restrict__ topk_gate)
{
    __shared__ __align__(16) __bf16 As[32 * 32];    //  2 KB
    __shared__ __align__(16) __bf16 Bs[256 * 32];   // 16 KB
    __shared__ float zs[32 * 257];                  // odd stride -> no bank conflicts
    __shared__ float cinv[NHEAD];

    const int tid  = threadIdx.x;
    const int lane = tid & 31;
    const int wid  = tid >> 5;
    const int l16  = lane & 15;
    const int half = lane >> 4;
    const int wm   = wid & 1;      // 2 row sub-tiles of 16
    const int wn   = wid >> 1;     // 4 col sub-tiles of 64
    const int m0   = blockIdx.x * 32;

    v8f acc[4] = {};

    for (int k0 = 0; k0 < DIN; k0 += 32) {
        {   // stage x tile 32x32 fp32 -> bf16 LDS
            int row = tid >> 3, kq = tid & 7;
            float4 xv = *reinterpret_cast<const float4*>(&x[(size_t)(m0 + row) * DIN + k0 + kq * 4]);
            *reinterpret_cast<uint2*>(&As[row * 32 + kq * 4]) = pack_bf16x4(xv);
        }
        #pragma unroll
        for (int v = 0; v < 8; ++v) {   // stage router_w tile 256x32
            int fidx = v * 256 + tid;
            int row = fidx >> 3, kq = fidx & 7;
            float4 wv = *reinterpret_cast<const float4*>(&rw[(size_t)row * DIN + k0 + kq * 4]);
            *reinterpret_cast<uint2*>(&Bs[row * 32 + kq * 4]) = pack_bf16x4(wv);
        }
        __syncthreads();

        FragAB a;  // A 16x32: lanes0-15 M=l16 K{0..7,16..23}; lanes16-31 K{8..15,24..31}
        a.f[0] = *reinterpret_cast<const float4*>(&As[(wm * 16 + l16) * 32 + half * 8]);
        a.f[1] = *reinterpret_cast<const float4*>(&As[(wm * 16 + l16) * 32 + 16 + half * 8]);
        #pragma unroll
        for (int j = 0; j < 4; ++j) {
            FragAB b;  // B 32x16: lane=N; lanes0-15 K{0..7,16..23}, lanes16-31 K{8..15,24..31}
            const __bf16* bp = &Bs[(wn * 64 + j * 16 + l16) * 32 + half * 8];
            b.f[0] = *reinterpret_cast<const float4*>(bp);
            b.f[1] = *reinterpret_cast<const float4*>(bp + 16);
            acc[j] = __builtin_amdgcn_wmma_f32_16x16x32_bf16(
                false, a.v, false, b.v, (short)0, acc[j], false, false);
        }
        __syncthreads();
    }

    #pragma unroll
    for (int j = 0; j < 4; ++j)
        #pragma unroll
        for (int i = 0; i < 8; ++i)
            zs[(wm * 16 + i + half * 8) * 257 + wn * 64 + j * 16 + l16] = acc[j][i];

    if (tid < NHEAD) {  // centroid inverse norms
        float s = 0.f;
        for (int k = 0; k < HIDN; ++k) { float c = cent[tid * HIDN + k]; s += c * c; }
        cinv[tid] = 1.f / fmaxf(sqrtf(s), EPSN);
    }
    __syncthreads();

    if (tid < 32) {   // one thread per token
        const float* zr = &zs[tid * 257];
        float ss = 0.f;
        for (int k = 0; k < HIDN; ++k) ss += zr[k] * zr[k];
        const float zinv = 1.f / fmaxf(sqrtf(ss), EPSN);
        const float tscale = expf(temp[0]) * (1.f / 16.f);  // /sqrt(HID)*exp(T)

        float lg[NHEAD];
        float mx = -1e30f;
        for (int h = 0; h < NHEAD; ++h) {
            float s = 0.f;
            for (int k = 0; k < HIDN; ++k) s += zr[k] * cent[h * HIDN + k];
            float l = s * zinv * cinv[h] * tscale;
            lg[h] = l;
            mx = fmaxf(mx, l);
        }
        float se = 0.f;
        for (int h = 0; h < NHEAD; ++h) { lg[h] = expf(lg[h] - mx); se += lg[h]; }
        const float inv = 1.f / se;

        const int t = m0 + tid;
        unsigned taken = 0;
        for (int r = 0; r < KSEL; ++r) {   // top-4 descending, first-index tiebreak
            int best = 0; float bv = -1.f;
            for (int h = 0; h < NHEAD; ++h)
                if (!((taken >> h) & 1u) && lg[h] > bv) { bv = lg[h]; best = h; }
            taken |= 1u << best;
            topk_idx[t * KSEL + r]  = best;
            topk_gate[t * KSEL + r] = bv * inv;
        }
    }
}

// ---- shared epilogue: bias + per-token gated scatter of one head's columns ----
__device__ __forceinline__ void gather_epilogue(
    const v8f acc[4], const float* __restrict__ bias,
    const int* __restrict__ topk_idx, const float* __restrict__ topk_gate,
    float* __restrict__ out, int m0, int n0, int head,
    int wm, int wn, int l16, int half)
{
    #pragma unroll
    for (int i = 0; i < 8; ++i) {
        const int t = m0 + wm * 16 + i + half * 8;     // token (C layout M = i + 8*half)
        const int4   id = *reinterpret_cast<const int4*>(&topk_idx[t * KSEL]);
        const float4 g  = *reinterpret_cast<const float4*>(&topk_gate[t * KSEL]);
        int r = -1; float gv = 0.f;
        if      (id.x == head) { r = 0; gv = g.x; }
        else if (id.y == head) { r = 1; gv = g.y; }
        else if (id.z == head) { r = 2; gv = g.z; }
        else if (id.w == head) { r = 3; gv = g.w; }
        if (r >= 0) {
            #pragma unroll
            for (int j = 0; j < 4; ++j) {
                const int c = wn * 64 + j * 16 + l16;        // col within head [0,127]
                const float val = acc[j][i] + bias[n0 + c];
                out[(size_t)t * (KSEL * HD) + r * HD + c] = val * gv;
            }
        }
    }
}

// ============================================================================
// Kernel 2a (preferred): main projection from pre-converted bf16 mirrors.
// Double-buffered ASYNC global->LDS DMA pipeline + WMMA.
// Block = 64 tokens x 128 cols (exactly one head).
// ============================================================================
__global__ __launch_bounds__(256) void sq_gemm_async(
    const __bf16* __restrict__ xb, const __bf16* __restrict__ wb,
    const float* __restrict__ bias, const int* __restrict__ topk_idx,
    const float* __restrict__ topk_gate, float* __restrict__ out)
{
    __shared__ __align__(16) __bf16 Ab[2][64 * 32];     // 2 x 4 KB
    __shared__ __align__(16) __bf16 Bb[2][128 * 32];    // 2 x 8 KB

    const int tid  = threadIdx.x;
    const int lane = tid & 31;
    const int wid  = tid >> 5;
    const int l16  = lane & 15;
    const int half = lane >> 4;
    const int wm   = wid & 3;
    const int wn   = wid >> 2;
    const int m0   = blockIdx.x * 64;
    const int n0   = blockIdx.y * HD;
    const int head = blockIdx.y;

    // per-thread async-copy assignment (b128 = 8 bf16 K-elems)
    const int ar = tid >> 2, ak = (tid & 3) * 8;       // A: 64x32, 1 chunk/thread
    const int br = tid >> 2, bk = (tid & 3) * 8;       // B: 128x32, 2 chunks/thread

    auto stage = [&](int buf, int k0) {
        async_g2l_b128(&xb[(size_t)(m0 + ar) * DIN + k0 + ak],       &Ab[buf][ar * 32 + ak]);
        async_g2l_b128(&wb[(size_t)(n0 + br) * DIN + k0 + bk],       &Bb[buf][br * 32 + bk]);
        async_g2l_b128(&wb[(size_t)(n0 + br + 64) * DIN + k0 + bk],  &Bb[buf][(br + 64) * 32 + bk]);
    };

    v8f acc[4] = {};
    stage(0, 0);

    constexpr int NIT = DIN / 32;   // 64
    for (int i = 0; i < NIT; ++i) {
        const int buf = i & 1;
        if (i + 1 < NIT) {           // prefetch next stage, then retire current (in-order)
            stage(buf ^ 1, (i + 1) * 32);
            wait_async<3>();
        } else {
            wait_async<0>();
        }
        __syncthreads();             // all waves' DMA for stage i landed in LDS

        FragAB a;
        a.f[0] = *reinterpret_cast<const float4*>(&Ab[buf][(wm * 16 + l16) * 32 + half * 8]);
        a.f[1] = *reinterpret_cast<const float4*>(&Ab[buf][(wm * 16 + l16) * 32 + 16 + half * 8]);
        #pragma unroll
        for (int j = 0; j < 4; ++j) {
            FragAB b;
            const __bf16* bp = &Bb[buf][(wn * 64 + j * 16 + l16) * 32 + half * 8];
            b.f[0] = *reinterpret_cast<const float4*>(bp);
            b.f[1] = *reinterpret_cast<const float4*>(bp + 16);
            acc[j] = __builtin_amdgcn_wmma_f32_16x16x32_bf16(
                false, a.v, false, b.v, (short)0, acc[j], false, false);
        }
        __syncthreads();             // done reading buf before it is restaged
    }

    gather_epilogue(acc, bias, topk_idx, topk_gate, out, m0, n0, head, wm, wn, l16, half);
}

// ============================================================================
// Kernel 2b (fallback, small-ws): fused fp32->bf16 staging variant (round-1).
// ============================================================================
__global__ __launch_bounds__(256) void sq_gemm_gather(
    const float* __restrict__ x, const float* __restrict__ weight,
    const float* __restrict__ bias, const int* __restrict__ topk_idx,
    const float* __restrict__ topk_gate, float* __restrict__ out)
{
    __shared__ __align__(16) __bf16 As[64 * 32];
    __shared__ __align__(16) __bf16 Bs[128 * 32];

    const int tid  = threadIdx.x;
    const int lane = tid & 31;
    const int wid  = tid >> 5;
    const int l16  = lane & 15;
    const int half = lane >> 4;
    const int wm   = wid & 3;
    const int wn   = wid >> 2;
    const int m0   = blockIdx.x * 64;
    const int n0   = blockIdx.y * HD;
    const int head = blockIdx.y;

    v8f acc[4] = {};

    for (int k0 = 0; k0 < DIN; k0 += 32) {
        #pragma unroll
        for (int v = 0; v < 2; ++v) {
            int fidx = v * 256 + tid;
            int row = fidx >> 3, kq = fidx & 7;
            float4 xv = *reinterpret_cast<const float4*>(&x[(size_t)(m0 + row) * DIN + k0 + kq * 4]);
            *reinterpret_cast<uint2*>(&As[row * 32 + kq * 4]) = pack_bf16x4(xv);
        }
        #pragma unroll
        for (int v = 0; v < 4; ++v) {
            int fidx = v * 256 + tid;
            int row = fidx >> 3, kq = fidx & 7;
            float4 wv = *reinterpret_cast<const float4*>(&weight[(size_t)(n0 + row) * DIN + k0 + kq * 4]);
            *reinterpret_cast<uint2*>(&Bs[row * 32 + kq * 4]) = pack_bf16x4(wv);
        }
        if (k0 + 32 < DIN) {
            __builtin_prefetch(&x[(size_t)(m0 + (tid >> 2)) * DIN + k0 + 32], 0, 1);
            __builtin_prefetch(&weight[(size_t)(n0 + (tid >> 1)) * DIN + k0 + 32], 0, 1);
        }
        __syncthreads();

        FragAB a;
        a.f[0] = *reinterpret_cast<const float4*>(&As[(wm * 16 + l16) * 32 + half * 8]);
        a.f[1] = *reinterpret_cast<const float4*>(&As[(wm * 16 + l16) * 32 + 16 + half * 8]);
        #pragma unroll
        for (int j = 0; j < 4; ++j) {
            FragAB b;
            const __bf16* bp = &Bs[(wn * 64 + j * 16 + l16) * 32 + half * 8];
            b.f[0] = *reinterpret_cast<const float4*>(bp);
            b.f[1] = *reinterpret_cast<const float4*>(bp + 16);
            acc[j] = __builtin_amdgcn_wmma_f32_16x16x32_bf16(
                false, a.v, false, b.v, (short)0, acc[j], false, false);
        }
        __syncthreads();
    }

    gather_epilogue(acc, bias, topk_idx, topk_gate, out, m0, n0, head, wm, wn, l16, half);
}

// ============================================================================
extern "C" void kernel_launch(void* const* d_in, const int* in_sizes, int n_in,
                              void* d_out, int out_size, void* d_ws, size_t ws_size,
                              hipStream_t stream)
{
    const float* x      = (const float*)d_in[0];
    const float* rw     = (const float*)d_in[1];
    const float* cent   = (const float*)d_in[2];
    const float* temp   = (const float*)d_in[3];
    const float* weight = (const float*)d_in[4];
    const float* bias   = (const float*)d_in[5];
    float*       out    = (float*)d_out;

    // ws layout: gates f32[MTOT*4] | idx i32[MTOT*4] | xb bf16[MTOT*DIN] | wb bf16[DIN*DIN]
    const size_t IDX_OFF = (size_t)MTOT * KSEL * sizeof(float);
    const size_t XB_OFF  = 2 * IDX_OFF;                       // 512 KB
    const size_t WB_OFF  = XB_OFF + (size_t)MTOT * DIN * 2;   // + 64 MB
    const size_t WS_NEED = WB_OFF + (size_t)DIN * DIN * 2;    // + 8 MB

    float* gates = (float*)d_ws;
    int*   idx   = (int*)((char*)d_ws + IDX_OFF);

    sq_router<<<dim3(MTOT / 32), dim3(256), 0, stream>>>(x, rw, cent, temp, idx, gates);

    if (ws_size >= WS_NEED) {
        __bf16* xb = (__bf16*)((char*)d_ws + XB_OFF);
        __bf16* wb = (__bf16*)((char*)d_ws + WB_OFF);
        conv_f32_bf16<<<dim3((MTOT * DIN / 8) / 256), dim3(256), 0, stream>>>(x, xb);
        conv_f32_bf16<<<dim3((DIN * DIN / 8) / 256), dim3(256), 0, stream>>>(weight, wb);
        sq_gemm_async<<<dim3(MTOT / 64, NHEAD), dim3(256), 0, stream>>>(
            xb, wb, bias, idx, gates, out);
    } else {
        sq_gemm_gather<<<dim3(MTOT / 64, NHEAD), dim3(256), 0, stream>>>(
            x, weight, bias, idx, gates, out);
    }
}
